// GNNEncoder_66365834658322
// MI455X (gfx1250) — matrix operand order repaired and checked by reading, not compiled, not run on hardware
//
#include <hip/hip_runtime.h>

// ---------------- problem constants (fixed by the reference) ----------------
constexpr int NN   = 50000;   // nodes
constexpr int EE   = 400000;  // edges
constexpr int FD   = 512;     // feature dim (IN == HID)
constexpr int NRSU = 1000;
constexpr int NVEH = NN - NRSU;
constexpr int SCAN_NB = (NN + 255) / 256;   // 196 scan blocks
constexpr int MPAD = 50048;                 // NN rounded up to 128 (391 blocks)

typedef _Float16 v8h  __attribute__((ext_vector_type(8)));
typedef _Float16 v16h __attribute__((ext_vector_type(16)));
typedef float    v8f  __attribute__((ext_vector_type(8)));
typedef int      gv4i __attribute__((vector_size(16)));   // matches builtin param type

#if defined(__gfx1250__) && __has_builtin(__builtin_amdgcn_global_load_async_to_lds_b128) && __has_builtin(__builtin_amdgcn_s_wait_asynccnt)
#define USE_ASYNC_LDS 1
typedef __attribute__((address_space(1))) gv4i* as1_v4i_p;
typedef __attribute__((address_space(3))) gv4i* as3_v4i_p;
#else
#define USE_ASYNC_LDS 0
#endif

// ---------------- degree / norm ----------------
__global__ void k_indeg_zero(int* __restrict__ indeg) {
    int i = blockIdx.x * blockDim.x + threadIdx.x;
    if (i < NN) indeg[i] = 0;
}
__global__ void k_indeg_count(const int* __restrict__ dst, int* __restrict__ indeg) {
    int e = blockIdx.x * blockDim.x + threadIdx.x;
    if (e < EE) atomicAdd(&indeg[dst[e]], 1);
}
__global__ void k_dinv(const int* __restrict__ indeg, float* __restrict__ dinv) {
    int i = blockIdx.x * blockDim.x + threadIdx.x;
    if (i < NN) dinv[i] = rsqrtf((float)(indeg[i] + 1));   // +1 self-loop, deg >= 1
}

// ---------------- exclusive scan of indeg -> rowstart (CSR by dst) ---------
__global__ void k_scan_block(const int* __restrict__ indeg, int* __restrict__ rowstart,
                             int* __restrict__ blockSums) {
    __shared__ int sh[256];
    int i = blockIdx.x * 256 + threadIdx.x;
    int v = (i < NN) ? indeg[i] : 0;
    sh[threadIdx.x] = v;
    __syncthreads();
#pragma unroll
    for (int off = 1; off < 256; off <<= 1) {
        int t = (threadIdx.x >= off) ? sh[threadIdx.x - off] : 0;
        __syncthreads();
        sh[threadIdx.x] += t;
        __syncthreads();
    }
    if (i < NN) rowstart[i] = sh[threadIdx.x] - v;          // exclusive
    if (threadIdx.x == 255) blockSums[blockIdx.x] = sh[255];
}
__global__ void k_scan_tops(const int* __restrict__ blockSums, int* __restrict__ blockOffs) {
    __shared__ int sh[256];
    int tid = threadIdx.x;
    int v = (tid < SCAN_NB) ? blockSums[tid] : 0;
    sh[tid] = v;
    __syncthreads();
#pragma unroll
    for (int off = 1; off < 256; off <<= 1) {
        int t = (tid >= off) ? sh[tid - off] : 0;
        __syncthreads();
        sh[tid] += t;
        __syncthreads();
    }
    if (tid < SCAN_NB) blockOffs[tid] = sh[tid] - v;        // exclusive
}
__global__ void k_scan_add(int* __restrict__ rowstart, const int* __restrict__ blockOffs,
                           int* __restrict__ cursor) {
    int i = blockIdx.x * 256 + threadIdx.x;
    if (i < NN) {
        int r = rowstart[i] + blockOffs[blockIdx.x];
        rowstart[i] = r;
        cursor[i]   = r;
    }
}
// bucket-fill CSR: esrc[pos] = src, ew[pos] = dinv[src]*dinv[dst]
__global__ void k_csr_fill(const int* __restrict__ src, const int* __restrict__ dst,
                           const float* __restrict__ dinv, int* __restrict__ cursor,
                           int* __restrict__ esrc, float* __restrict__ ew) {
    int e = blockIdx.x * blockDim.x + threadIdx.x;
    if (e < EE) {
        int s = src[e], d = dst[e];
        int pos = atomicAdd(&cursor[d], 1);
        esrc[pos] = s;
        ew[pos]   = dinv[s] * dinv[d];
    }
}

// ---------------- conversions ----------------
__global__ void k_cvt_f16(const float* __restrict__ in, _Float16* __restrict__ out) {
    size_t idx = (size_t)blockIdx.x * blockDim.x + threadIdx.x;   // exactly NN*FD threads
    out[idx] = (_Float16)in[idx];
}
__global__ void k_zero_f16(_Float16* __restrict__ p, int n) {
    int i = blockIdx.x * blockDim.x + threadIdx.x;
    if (i < n) p[i] = (_Float16)0.0f;
}
// W (K x N row-major) -> WT (N x K row-major, f16)
__global__ void k_wt(const float* __restrict__ w, _Float16* __restrict__ wt) {
    int idx = blockIdx.x * 256 + threadIdx.x;       // FD*FD threads
    int n = idx >> 9, k = idx & (FD - 1);
    wt[(size_t)n * FD + k] = (_Float16)w[(size_t)k * FD + n];
}

// ---------------- WMMA GEMM:  Y(f16) = A(f16, M x K) * BT(f16, N x K)^T ----
// grid = (MPAD/128, FD/64), block = 256 (8 waves).
// Block computes a 128x64 output tile: wave w owns M-tile mBase = bx*128 + w*16;
// all 8 waves share a 64-row strip of BT staged once in LDS (64 KB, async copy).
__device__ __forceinline__ v16h ld_a_frag(const _Float16* p, int half) {
    v8h a0 = *(const v8h*)(p + half * 8);            // K: half*8    .. +7
    v8h a1 = *(const v8h*)(p + 16 + half * 8);       // K: 16+half*8 .. +7
    return __builtin_shufflevector(a0, a1, 0,1,2,3,4,5,6,7,8,9,10,11,12,13,14,15);
}
__device__ __forceinline__ v16h ld_b_frag(const _Float16* p, int half) {
    v8h b0 = *(const v8h*)(p + half * 8);            // K: half*8    .. +7
    v8h b1 = *(const v8h*)(p + 16 + half * 8);       // K: 16+half*8 .. +7
    return __builtin_shufflevector(b0, b1, 0,1,2,3,4,5,6,7,8,9,10,11,12,13,14,15);
}
__global__ __launch_bounds__(256) void k_gemm_wmma(const _Float16* __restrict__ A,
                                                   const _Float16* __restrict__ BT,
                                                   _Float16* __restrict__ Y) {
    __shared__ _Float16 ldsB[64 * FD];               // 64 KB of the 320 KB WGP LDS

    const int lane  = threadIdx.x & 31;
    const int wave  = threadIdx.x >> 5;
    const int m16   = lane & 15;
    const int half  = lane >> 4;
    const int mBase = blockIdx.x * 128 + wave * 16;
    const int nBase = blockIdx.y * 64;

    // ---- stage BT[nBase .. nBase+63][0..511] into LDS (linear 64 KB copy) ----
    const _Float16* gB = BT + (size_t)nBase * FD;
#if USE_ASYNC_LDS
    for (int c = threadIdx.x; c < (64 * FD) / 8; c += 256) {
        const void* gp = (const void*)(gB + (size_t)c * 8);
        void*       lp = (void*)(ldsB + (size_t)c * 8);
        __builtin_amdgcn_global_load_async_to_lds_b128((as1_v4i_p)gp, (as3_v4i_p)lp, 0, 0);
    }
    __builtin_amdgcn_s_wait_asynccnt(0);
#else
    for (int c = threadIdx.x; c < (64 * FD) / 8; c += 256) {
        *(v8h*)(ldsB + (size_t)c * 8) = *(const v8h*)(gB + (size_t)c * 8);
    }
#endif
    __syncthreads();

    const _Float16* pa  = A + (size_t)(mBase + m16) * FD;
    const _Float16* lb0 = ldsB + (size_t)( 0 + m16) * FD;
    const _Float16* lb1 = ldsB + (size_t)(16 + m16) * FD;
    const _Float16* lb2 = ldsB + (size_t)(32 + m16) * FD;
    const _Float16* lb3 = ldsB + (size_t)(48 + m16) * FD;

    v8f acc0 = {}, acc1 = {}, acc2 = {}, acc3 = {};
    v16h aCur = ld_a_frag(pa, half);                 // double-buffered A fragment
    for (int k = 0; k < FD; k += 32) {
        v16h aNext = aCur;
        if (k + 32 < FD) aNext = ld_a_frag(pa + k + 32, half);
        acc0 = __builtin_amdgcn_wmma_f32_16x16x32_f16(false, aCur, false,
                   ld_b_frag(lb0 + k, half), (short)0, acc0, false, false);
        acc1 = __builtin_amdgcn_wmma_f32_16x16x32_f16(false, aCur, false,
                   ld_b_frag(lb1 + k, half), (short)0, acc1, false, false);
        acc2 = __builtin_amdgcn_wmma_f32_16x16x32_f16(false, aCur, false,
                   ld_b_frag(lb2 + k, half), (short)0, acc2, false, false);
        acc3 = __builtin_amdgcn_wmma_f32_16x16x32_f16(false, aCur, false,
                   ld_b_frag(lb3 + k, half), (short)0, acc3, false, false);
        aCur = aNext;
    }
    // C/D layout: VGPR v -> row (v + half*8), col = m16
    _Float16* py = Y + (size_t)(mBase + half * 8) * FD + nBase + m16;
#pragma unroll
    for (int v = 0; v < 8; ++v) {
        py[(size_t)v * FD +  0] = (_Float16)acc0[v];
        py[(size_t)v * FD + 16] = (_Float16)acc1[v];
        py[(size_t)v * FD + 32] = (_Float16)acc2[v];
        py[(size_t)v * FD + 48] = (_Float16)acc3[v];
    }
}

// ---------------- pull aggregation over dst-CSR ----------------------------
// agg[i] = y[i]*dinv[i]^2 + bias + sum_{e: dst=i} w_e * y[src_e]
// RELU16: write f16(relu(.)) to hOut (layer 1); else f32 to aggOut (layer 2)
template <bool RELU16>
__global__ __launch_bounds__(256) void k_aggregate(const int* __restrict__ rowstart,
                                                   const int* __restrict__ indeg,
                                                   const int* __restrict__ esrc,
                                                   const float* __restrict__ ew,
                                                   const _Float16* __restrict__ yh,
                                                   const float* __restrict__ dinv,
                                                   const float* __restrict__ bias,
                                                   float* __restrict__ aggOut,
                                                   _Float16* __restrict__ hOut) {
    const int i = blockIdx.x;                   // one block per node
    const int f = threadIdx.x;                  // features f and f+256
    const size_t base = (size_t)i * FD;
    const float di = dinv[i];
    float acc0 = (float)yh[base + f]       * di * di + bias[f];
    float acc1 = (float)yh[base + f + 256] * di * di + bias[f + 256];

    const int jb = rowstart[i];
    const int je = jb + indeg[i];
    for (int j = jb; j < je; ++j) {
        if (j + 1 < je)                                         // gfx1250 global_prefetch
            __builtin_prefetch(yh + (size_t)esrc[j + 1] * FD + f, 0, 1);
        const _Float16* py = yh + (size_t)esrc[j] * FD;
        const float w = ew[j];
        acc0 += w * (float)py[f];
        acc1 += w * (float)py[f + 256];
    }
    if (RELU16) {
        hOut[base + f]       = (_Float16)(acc0 > 0.0f ? acc0 : 0.0f);
        hOut[base + f + 256] = (_Float16)(acc1 > 0.0f ? acc1 : 0.0f);
    } else {
        aggOut[base + f]       = acc0;
        aggOut[base + f + 256] = acc1;
    }
}

// ---------------- pooling ----------------
__global__ void k_zero(float* __restrict__ p, int n) {
    int i = blockIdx.x * blockDim.x + threadIdx.x;
    if (i < n) p[i] = 0.0f;
}
__global__ __launch_bounds__(256) void k_pool_scatter(const float* __restrict__ agg,
                                                      const int* __restrict__ batch,
                                                      float* __restrict__ s,
                                                      float* __restrict__ cnt) {
    int node = NRSU + blockIdx.x;                 // NVEH blocks
    int seg  = batch[node];
    const float* pa = agg + (size_t)node * FD;
    float* ps = s + (size_t)seg * FD;
    int f = threadIdx.x;
    atomicAdd(&ps[f],       pa[f]);
    atomicAdd(&ps[f + 256], pa[f + 256]);
    if (f == 0) atomicAdd(&cnt[seg], 1.0f);
}
__global__ void k_finalize(const float* __restrict__ agg, const float* __restrict__ s,
                           const float* __restrict__ cnt, float* __restrict__ out) {
    size_t idx = (size_t)blockIdx.x * blockDim.x + threadIdx.x;   // NRSU*FD threads
    int r = (int)(idx >> 9), f = (int)(idx & (FD - 1));
    out[(size_t)r * 1024 + f] = agg[idx];
    float c = cnt[r]; c = c > 1.0f ? c : 1.0f;
    out[(size_t)r * 1024 + 512 + f] = s[idx] / c;
}

// ---------------- launcher ----------------
extern "C" void kernel_launch(void* const* d_in, const int* in_sizes, int n_in,
                              void* d_out, int out_size, void* d_ws, size_t ws_size,
                              hipStream_t stream) {
    const float* x    = (const float*)d_in[0];
    const int*   eidx = (const int*)  d_in[1];   // [0..E) = src, [E..2E) = dst
    const int*   bat  = (const int*)  d_in[2];
    const float* W1   = (const float*)d_in[3];
    const float* b1   = (const float*)d_in[4];
    const float* W2   = (const float*)d_in[5];
    const float* b2   = (const float*)d_in[6];
    float* out = (float*)d_out;

    const int* src = eidx;
    const int* dst = eidx + EE;

    // workspace carve-out (256B aligned)
    char* ws = (char*)d_ws;
    size_t off = 0;
    auto carve = [&](size_t bytes) { char* p = ws + off; off = (off + bytes + 255) & ~(size_t)255; return p; };
    int*      indeg   = (int*)     carve((size_t)NN * 4);
    int*      rowstart= (int*)     carve((size_t)NN * 4);
    int*      cursor  = (int*)     carve((size_t)NN * 4);
    int*      bsums   = (int*)     carve((size_t)256 * 4);
    int*      boffs   = (int*)     carve((size_t)256 * 4);
    int*      esrc    = (int*)     carve((size_t)EE * 4);
    float*    ew      = (float*)   carve((size_t)EE * 4);
    float*    dinv    = (float*)   carve((size_t)NN * 4);
    _Float16* xh      = (_Float16*)carve((size_t)MPAD * FD * 2);   // x_h, later h_h
    _Float16* wt      = (_Float16*)carve((size_t)FD * FD * 2);     // W1T then W2T
    _Float16* yh      = (_Float16*)carve((size_t)MPAD * FD * 2);   // GEMM output (f16)
    float*    agg     = (float*)   carve((size_t)NN * FD * 4);     // layer-2 node output
    float*    spool   = (float*)   carve((size_t)NRSU * FD * 4);
    float*    cnt     = (float*)   carve((size_t)NRSU * 4);
    (void)ws_size; (void)in_sizes; (void)n_in; (void)out_size;

    const int elemBlocks = (NN * FD) / 256;          // 100000, exact
    const dim3 gemmGrid(MPAD / 128, FD / 64);        // (391, 8), exact -> EXEC all ones
    const int padElems = (MPAD - NN) * FD;           // 24576 pad f16 elements

    // ---- graph preprocessing (CSR by dst, built once, used both layers) ----
    k_indeg_zero <<<SCAN_NB, 256, 0, stream>>>(indeg);
    k_indeg_count<<<(EE + 255) / 256, 256, 0, stream>>>(dst, indeg);
    k_dinv       <<<SCAN_NB, 256, 0, stream>>>(indeg, dinv);
    k_scan_block <<<SCAN_NB, 256, 0, stream>>>(indeg, rowstart, bsums);
    k_scan_tops  <<<1, 256, 0, stream>>>(bsums, boffs);
    k_scan_add   <<<SCAN_NB, 256, 0, stream>>>(rowstart, boffs, cursor);
    k_csr_fill   <<<(EE + 255) / 256, 256, 0, stream>>>(src, dst, dinv, cursor, esrc, ew);

    // ---- layer 1 ----
    k_cvt_f16<<<elemBlocks, 256, 0, stream>>>(x, xh);
    k_zero_f16<<<(padElems + 255) / 256, 256, 0, stream>>>(xh + (size_t)NN * FD, padElems);
    k_wt     <<<(FD * FD) / 256, 256, 0, stream>>>(W1, wt);
    k_gemm_wmma<<<gemmGrid, 256, 0, stream>>>(xh, wt, yh);
    k_aggregate<true><<<NN, 256, 0, stream>>>(rowstart, indeg, esrc, ew, yh, dinv, b1,
                                              nullptr, xh);       // h_h reuses xh (pad rows stay 0)

    // ---- layer 2 ----
    k_wt     <<<(FD * FD) / 256, 256, 0, stream>>>(W2, wt);
    k_gemm_wmma<<<gemmGrid, 256, 0, stream>>>(xh, wt, yh);
    k_aggregate<false><<<NN, 256, 0, stream>>>(rowstart, indeg, esrc, ew, yh, dinv, b2,
                                               agg, nullptr);

    // ---- pooling + output ----
    k_zero<<<(NRSU * FD + 255) / 256, 256, 0, stream>>>(spool, NRSU * FD);
    k_zero<<<(NRSU + 255) / 256, 256, 0, stream>>>(cnt, NRSU);
    k_pool_scatter<<<NVEH, 256, 0, stream>>>(agg, bat, spool, cnt);
    k_finalize<<<(NRSU * FD) / 256, 256, 0, stream>>>(agg, spool, cnt, out);
}
